// GraphEmbLayer_46643344835307
// MI455X (gfx1250) — matrix operand order repaired and checked by reading, not compiled, not run on hardware
//
#include <hip/hip_runtime.h>
#include <hip/hip_bf16.h>
#include <math.h>

// ---------------------------------------------------------------------------
// CDNA5 (gfx1250) WMMA types
// ---------------------------------------------------------------------------
typedef __bf16 bf16_t;
typedef __attribute__((ext_vector_type(16))) __bf16 v16bf;
typedef __attribute__((ext_vector_type(8)))  float  v8f;
typedef __attribute__((ext_vector_type(4)))  unsigned int u32x4;
typedef __attribute__((ext_vector_type(8)))  int  i32x8;
typedef __attribute__((ext_vector_type(4)))  int  i32x4;

constexpr int Dm = 192;   // event dim (12 tiles of 16)
constexpr int Lg = 128;   // events per graph (8 tiles of 16)

// Tensor Data Mover availability (arity differs per toolchain; see bridge doc)
#if defined(__has_builtin)
#  if __has_builtin(__builtin_amdgcn_tensor_load_to_lds) && \
      __has_builtin(__builtin_amdgcn_s_wait_tensorcnt)
#    define HAVE_TDM 1
#  endif
#endif
#ifndef HAVE_TDM
#  define HAVE_TDM 0
#endif

// ---------------------------------------------------------------------------
// TDM: async 2D bf16 tile load (global -> LDS), D# per ISA 08_async_tensor §8.
// Issue from ONE wave only (TDM ignores EXEC: gate at wave granularity).
// ---------------------------------------------------------------------------
__device__ __forceinline__ void tdm_load_2d_bf16(const bf16_t* gsrc, void* ldst,
                                                 unsigned int dim0, unsigned int dim1) {
#if HAVE_TDM
  const unsigned long long ga = (unsigned long long)(uintptr_t)gsrc;
  const unsigned int la = (unsigned int)(uintptr_t)ldst;  // flat[31:0] == LDS offset
  u32x4 g0; i32x8 g1; i32x4 g2; i32x4 g3;
  g0[0] = 1u;                                  // count=1, user descriptor
  g0[1] = la;                                  // lds_addr
  g0[2] = (unsigned int)ga;                    // global_addr[31:0]
  g0[3] = (unsigned int)((ga >> 32) & 0x01FFFFFFull) | (2u << 30);  // [120:64], type=2
  g1[0] = (int)(1u << 16);                     // workgroup_mask=0, data_size=1 (2B)
  g1[1] = (int)((dim0 & 0xFFFFu) << 16);       // tensor_dim0[15:0] @ bits 63:48
  g1[2] = (int)(((dim0 >> 16) & 0xFFFFu) | ((dim1 & 0xFFFFu) << 16));   // dim0 hi, dim1 lo
  g1[3] = (int)(((dim1 >> 16) & 0xFFFFu) | ((dim0 & 0xFFFFu) << 16));   // dim1 hi, tile_dim0
  g1[4] = (int)(dim1 & 0xFFFFu);               // tile_dim1 (tile_dim2=0)
  g1[5] = (int)dim0;                           // tensor_dim0_stride[31:0]
  g1[6] = 0; g1[7] = 0;                        // stride hi, tensor_dim1_stride=0
  g2[0] = g2[1] = g2[2] = g2[3] = 0;
  g3[0] = g3[1] = g3[2] = g3[3] = 0;
#  if __clang_major__ >= 23
  i32x8 g4; g4[0]=g4[1]=g4[2]=g4[3]=g4[4]=g4[5]=g4[6]=g4[7]=0;
  __builtin_amdgcn_tensor_load_to_lds(g0, g1, g2, g3, g4, 0);
#  else
  __builtin_amdgcn_tensor_load_to_lds(g0, g1, g2, g3, 0);
#  endif
#else
  (void)gsrc; (void)ldst; (void)dim0; (void)dim1;
#endif
}

__device__ __forceinline__ void tdm_wait() {
#if HAVE_TDM
  __builtin_amdgcn_s_wait_tensorcnt(0);
#endif
}

// ---------------------------------------------------------------------------
// Fragment loaders (row/col per ISA 7.12.2, contiguous k = 16h + i, wave32)
// ---------------------------------------------------------------------------
// A-fragment (16x32 bf16): lane half h, row m = lane&15.
// Per lane: 8 dword pairs at k/2 = h*8 + j (element i <-> K=h*16+i).
__device__ __forceinline__ v16bf afrag_lds(const bf16_t* rowk, int h) {
  union { v16bf v; unsigned int u[8]; } a;
  const unsigned int* p = (const unsigned int*)rowk;
  const int o = h * 8;
#pragma unroll
  for (int j = 0; j < 4; ++j) {
    a.u[j]     = p[o + j];
    a.u[4 + j] = p[o + 4 + j];
  }
  return a.v;
}

// B-fragment (32x16 bf16): lane holds column n = lane&15; element i <-> K=h*16+i
// -> 16 contiguous bf16 per lane (works for LDS or global bf16).
__device__ __forceinline__ v16bf bfrag16(const bf16_t* p) {
  union { v16bf v; uint4 q[2]; } b;
  b.q[0] = ((const uint4*)p)[0];
  b.q[1] = ((const uint4*)p)[1];
  return b.v;
}

__device__ __forceinline__ v8f wmma_bf16(v16bf a, v16bf b, v8f c) {
  return __builtin_amdgcn_wmma_f32_16x16x32_bf16(false, a, false, b,
                                                 (short)0, c, false, false);
}

// ---------------------------------------------------------------------------
// Kernel 0: one-shot fp32 -> bf16 weight conversion (L2-resident afterwards)
// ---------------------------------------------------------------------------
__global__ __launch_bounds__(256)
void cvt_weights_kernel(const float* __restrict__ WK, const float* __restrict__ WQ,
                        const float* __restrict__ Wgate, const float* __restrict__ Wg,
                        bf16_t* __restrict__ WKb, bf16_t* __restrict__ WQb,
                        bf16_t* __restrict__ WGateb, bf16_t* __restrict__ WGb) {
  const int i = blockIdx.x * 256 + threadIdx.x;
  if (i < Dm * Dm) {
    WKb[i] = (__bf16)WK[i];
    WQb[i] = (__bf16)WQ[i];
    WGb[i] = (__bf16)Wg[i];
  }
  if (i < 2 * Dm * Dm) WGateb[i] = (__bf16)Wgate[i];
}

// ---------------------------------------------------------------------------
// Kernel 1: K = E @ Wk^T, Q = E @ Wq^T  (bf16 out, f32 accumulate)
// ---------------------------------------------------------------------------
__global__ __launch_bounds__(256)
void proj_kq_kernel(const float* __restrict__ E, const bf16_t* __restrict__ WKb,
                    const bf16_t* __restrict__ WQb, bf16_t* __restrict__ Kout,
                    bf16_t* __restrict__ Qout) {
  __shared__ bf16_t sA[Lg * Dm];               // 48 KB
  const int tid  = threadIdx.x;
  const int row0 = blockIdx.x * Lg;

  for (int i = tid; i < Lg * Dm; i += 256)     // fp32 -> bf16 while staging
    sA[i] = (__bf16)E[(size_t)row0 * Dm + i];
  __syncthreads();

  const int w = tid >> 5, l = tid & 31, nl = l & 15, h = l >> 4;
  const int m0 = w * 16;

  v16bf af[6];
#pragma unroll
  for (int kk = 0; kk < 6; ++kk)
    af[kk] = afrag_lds(&sA[(m0 + nl) * Dm + kk * 32], h);

  for (int t = 0; t < 12; ++t) {
    const int n = t * 16 + nl;                 // B[k][n] = W[n][k] (torch W^T)
    if (t + 1 < 12) {                          // global_prefetch next weight rows
      __builtin_prefetch(&WKb[(size_t)(n + 16) * Dm], 0, 3);
      __builtin_prefetch(&WQb[(size_t)(n + 16) * Dm], 0, 3);
    }
    v8f accK = {}; v8f accQ = {};
#pragma unroll
    for (int kk = 0; kk < 6; ++kk) {
      v16bf bk = bfrag16(&WKb[(size_t)n * Dm + kk * 32 + h * 16]);
      accK = wmma_bf16(af[kk], bk, accK);
      v16bf bq = bfrag16(&WQb[(size_t)n * Dm + kk * 32 + h * 16]);
      accQ = wmma_bf16(af[kk], bq, accQ);
    }
#pragma unroll
    for (int r = 0; r < 8; ++r) {              // C/D: row = r + 8h, col = n
      const size_t o = (size_t)(row0 + m0 + r + 8 * h) * Dm + n;
      Kout[o] = (__bf16)accK[r];
      Qout[o] = (__bf16)accQ[r];
    }
  }
}

// ---------------------------------------------------------------------------
// Kernel 2: per-graph attention, fully in LDS (240 KB dynamic).
//   S = Qg Kg^T / sqrt(D); softmax rows; att_emb = att @ Eg  -> bf16 out
// ---------------------------------------------------------------------------
__global__ __launch_bounds__(256)
void attn_kernel(const float* __restrict__ E, const bf16_t* __restrict__ Kbf,
                 const bf16_t* __restrict__ Qbf, bf16_t* __restrict__ AEout) {
  extern __shared__ char smem[];
  bf16_t* sQ   = (bf16_t*)smem;                // 128*192 bf16  (48 KB)
  bf16_t* sK   = sQ + Lg * Dm;                 // 128*192 bf16  (48 KB)
  bf16_t* sEt  = sK + Lg * Dm;                 // 192*128 bf16, transposed (48 KB)
  float*  sS   = (float*)(sEt + Dm * Lg);      // 128*128 f32   (64 KB)
  bf16_t* sAtt = (bf16_t*)(sS + Lg * Lg);      // 128*128 bf16  (32 KB)

  const int g = blockIdx.x, tid = threadIdx.x;
  const size_t base = (size_t)g * Lg * Dm;

#if HAVE_TDM
  if (tid < 32) {                              // wave 0 issues both DMAs
    tdm_load_2d_bf16(Qbf + base, sQ, Dm, Lg);
    tdm_load_2d_bf16(Kbf + base, sK, Dm, Lg);
  }
#else
  {
    const uint4* gq = (const uint4*)(Qbf + base);
    const uint4* gk = (const uint4*)(Kbf + base);
    uint4* q4 = (uint4*)sQ; uint4* k4 = (uint4*)sK;
    for (int i = tid; i < Lg * Dm / 8; i += 256) { q4[i] = gq[i]; k4[i] = gk[i]; }
  }
#endif
  for (int i = tid; i < Lg * Dm; i += 256) {   // stage E transposed (B operand)
    const int k = i / Dm, n = i - k * Dm;
    sEt[n * Lg + k] = (__bf16)E[base + i];
  }
#if HAVE_TDM
  if (tid < 32) tdm_wait();
#endif
  __syncthreads();

  const int w = tid >> 5, l = tid & 31, nl = l & 15, h = l >> 4;
  const int m0 = w * 16;
  const float scale = 0.0721687836487032f;     // 1/sqrt(192)

  {                                            // scores: wave w -> rows m0..m0+15
    v16bf af[6];
#pragma unroll
    for (int kk = 0; kk < 6; ++kk)
      af[kk] = afrag_lds(&sQ[(m0 + nl) * Dm + kk * 32], h);
    for (int tn = 0; tn < 8; ++tn) {
      v8f acc = {};
#pragma unroll
      for (int kk = 0; kk < 6; ++kk) {         // B[k][n] = Kg[n][k] -> contiguous
        v16bf b = bfrag16(&sK[(tn * 16 + nl) * Dm + kk * 32 + h * 16]);
        acc = wmma_bf16(af[kk], b, acc);
      }
#pragma unroll
      for (int r = 0; r < 8; ++r)
        sS[(m0 + r + 8 * h) * Lg + tn * 16 + nl] = acc[r] * scale;
    }
  }
  __syncthreads();

  if (tid < Lg) {                              // row softmax over keys
    float mx = -3.4e38f;
    for (int j = 0; j < Lg; ++j) mx = fmaxf(mx, sS[tid * Lg + j]);
    float sum = 0.f;
    for (int j = 0; j < Lg; ++j) {
      const float e = __expf(sS[tid * Lg + j] - mx);
      sS[tid * Lg + j] = e; sum += e;
    }
    const float inv = 1.f / sum;
    for (int j = 0; j < Lg; ++j) sAtt[tid * Lg + j] = (__bf16)(sS[tid * Lg + j] * inv);
  }
  __syncthreads();

  {                                            // att_emb = att @ Eg (K = 128)
    v16bf af[4];
#pragma unroll
    for (int kk = 0; kk < 4; ++kk)
      af[kk] = afrag_lds(&sAtt[(m0 + nl) * Lg + kk * 32], h);
    for (int tc = 0; tc < 12; ++tc) {
      v8f acc = {};
#pragma unroll
      for (int kk = 0; kk < 4; ++kk) {         // B[k][n] = Eg[k][n] = sEt[n][k]
        v16bf b = bfrag16(&sEt[(tc * 16 + nl) * Lg + kk * 32 + h * 16]);
        acc = wmma_bf16(af[kk], b, acc);
      }
#pragma unroll
      for (int r = 0; r < 8; ++r)
        AEout[base + (size_t)(m0 + r + 8 * h) * Dm + tc * 16 + nl] = (__bf16)acc[r];
    }
  }
}

// ---------------------------------------------------------------------------
// Kernel 3: gated MLP (gate_in = att_emb @ Wgate^T + b) + mean pool per graph
// ---------------------------------------------------------------------------
__global__ __launch_bounds__(256)
void gate_pool_kernel(const bf16_t* __restrict__ AE, const bf16_t* __restrict__ WGateb,
                      const float* __restrict__ bgate, bf16_t* __restrict__ Pbf) {
  extern __shared__ char smem[];
  bf16_t* sA = (bf16_t*)smem;                  // 128*192 bf16 (48 KB)
  float*  sG = (float*)(sA + Lg * Dm);         // 128*384 f32 (192 KB)

  const int g = blockIdx.x, tid = threadIdx.x;
  const size_t base = (size_t)g * Lg * Dm;
#if HAVE_TDM
  if (tid < 32) {
    tdm_load_2d_bf16(AE + base, sA, Dm, Lg);
    tdm_wait();
  }
#else
  {
    const uint4* ga = (const uint4*)(AE + base);
    uint4* a4 = (uint4*)sA;
    for (int i = tid; i < Lg * Dm / 8; i += 256) a4[i] = ga[i];
  }
#endif
  __syncthreads();

  const int w = tid >> 5, l = tid & 31, nl = l & 15, h = l >> 4;
  const int m0 = w * 16;
  v16bf af[6];
#pragma unroll
  for (int kk = 0; kk < 6; ++kk)
    af[kk] = afrag_lds(&sA[(m0 + nl) * Dm + kk * 32], h);

  for (int tj = 0; tj < 24; ++tj) {            // 384 output cols
    const int j = tj * 16 + nl;
    if (tj + 1 < 24) __builtin_prefetch(&WGateb[(size_t)(j + 16) * Dm], 0, 3);
    const float bias = bgate[j];
    v8f acc = {};
#pragma unroll
    for (int kk = 0; kk < 6; ++kk) {
      v16bf b = bfrag16(&WGateb[(size_t)j * Dm + kk * 32 + h * 16]);
      acc = wmma_bf16(af[kk], b, acc);
    }
#pragma unroll
    for (int r = 0; r < 8; ++r)
      sG[(m0 + r + 8 * h) * 384 + j] = acc[r] + bias;
  }
  __syncthreads();

  if (tid < Dm) {                              // ev = y * sigmoid(x); mean over rows
    float s = 0.f;
#pragma unroll 4
    for (int r = 0; r < Lg; ++r) {
      const float x = sG[r * 384 + tid];
      const float y = sG[r * 384 + Dm + tid];
      s += y * (1.f / (1.f + __expf(-x)));
    }
    Pbf[(size_t)g * Dm + tid] = (__bf16)(s * (1.0f / (float)Lg));
  }
}

// ---------------------------------------------------------------------------
// Kernel 4: out = pooled @ Wg^T + bg   (512x192 @ 192x192, fp32 out)
// ---------------------------------------------------------------------------
__global__ __launch_bounds__(256)
void final_kernel(const bf16_t* __restrict__ Pbf, const bf16_t* __restrict__ WGb,
                  const float* __restrict__ bg, float* __restrict__ Out) {
  __shared__ bf16_t sP[Lg * Dm];               // 48 KB
  const int tid = threadIdx.x;
  const int row0 = blockIdx.x * Lg;
#if HAVE_TDM
  if (tid < 32) {
    tdm_load_2d_bf16(Pbf + (size_t)row0 * Dm, sP, Dm, Lg);
    tdm_wait();
  }
#else
  {
    const uint4* gp = (const uint4*)(Pbf + (size_t)row0 * Dm);
    uint4* p4 = (uint4*)sP;
    for (int i = tid; i < Lg * Dm / 8; i += 256) p4[i] = gp[i];
  }
#endif
  __syncthreads();

  const int w = tid >> 5, l = tid & 31, nl = l & 15, h = l >> 4;
  const int m0 = w * 16;
  v16bf af[6];
#pragma unroll
  for (int kk = 0; kk < 6; ++kk)
    af[kk] = afrag_lds(&sP[(m0 + nl) * Dm + kk * 32], h);

  for (int t = 0; t < 12; ++t) {
    const int n = t * 16 + nl;
    const float bias = bg[n];
    v8f acc = {};
#pragma unroll
    for (int kk = 0; kk < 6; ++kk) {
      v16bf b = bfrag16(&WGb[(size_t)n * Dm + kk * 32 + h * 16]);
      acc = wmma_bf16(af[kk], b, acc);
    }
#pragma unroll
    for (int r = 0; r < 8; ++r)
      Out[(size_t)(row0 + m0 + r + 8 * h) * Dm + n] = acc[r] + bias;
  }
}

// ---------------------------------------------------------------------------
// Host launcher
// ---------------------------------------------------------------------------
extern "C" void kernel_launch(void* const* d_in, const int* in_sizes, int n_in,
                              void* d_out, int out_size, void* d_ws, size_t ws_size,
                              hipStream_t stream) {
  const float* E     = (const float*)d_in[0];
  const float* WK    = (const float*)d_in[1];
  const float* WQ    = (const float*)d_in[2];
  const float* Wgate = (const float*)d_in[3];
  const float* bgate = (const float*)d_in[4];
  const float* Wg    = (const float*)d_in[5];
  const float* bg    = (const float*)d_in[6];
  float* Out = (float*)d_out;

  const int N = in_sizes[0] / Dm;              // 65536 events
  const int G = N / Lg;                        // 512 graphs

  // workspace carve-up (all buffers fully written before read)
  char* ws = (char*)d_ws;
  const size_t szNK = (size_t)N * Dm * sizeof(bf16_t);
  bf16_t* Kbf    = (bf16_t*)(ws);
  bf16_t* Qbf    = (bf16_t*)(ws + szNK);
  bf16_t* AEbf   = (bf16_t*)(ws + 2 * szNK);
  bf16_t* Pbf    = (bf16_t*)(ws + 3 * szNK);
  char*   wtap   = ws + 3 * szNK + (size_t)G * Dm * sizeof(bf16_t);
  wtap = (char*)(((uintptr_t)wtap + 255) & ~(uintptr_t)255);
  bf16_t* WKb    = (bf16_t*)(wtap);
  bf16_t* WQb    = (bf16_t*)(wtap + 1 * (size_t)Dm * Dm * 2);
  bf16_t* WGb    = (bf16_t*)(wtap + 2 * (size_t)Dm * Dm * 2);
  bf16_t* WGateb = (bf16_t*)(wtap + 3 * (size_t)Dm * Dm * 2);

  const int attnSmem = Lg * Dm * 2 * 3 + Lg * Lg * 4 + Lg * Lg * 2;   // 245760 B
  const int gateSmem = Lg * Dm * 2 + Lg * 384 * 4;                    // 245760 B
  (void)hipFuncSetAttribute((const void*)attn_kernel,
      hipFuncAttributeMaxDynamicSharedMemorySize, attnSmem);
  (void)hipFuncSetAttribute((const void*)gate_pool_kernel,
      hipFuncAttributeMaxDynamicSharedMemorySize, gateSmem);

  cvt_weights_kernel<<<(2 * Dm * Dm + 255) / 256, 256, 0, stream>>>(
      WK, WQ, Wgate, Wg, WKb, WQb, WGateb, WGb);
  proj_kq_kernel<<<N / Lg, 256, 0, stream>>>(E, WKb, WQb, Kbf, Qbf);
  attn_kernel<<<G, 256, attnSmem, stream>>>(E, Kbf, Qbf, AEbf);
  gate_pool_kernel<<<G, 256, gateSmem, stream>>>(AEbf, WGateb, bgate, Pbf);
  final_kernel<<<(G + Lg - 1) / Lg, 256, 0, stream>>>(Pbf, WGb, bg, Out);
}
